// ConformerEncoderLayer_19851338842234
// MI455X (gfx1250) — matrix operand order repaired and checked by reading, not compiled, not run on hardware
//
#include <hip/hip_runtime.h>

// ---------------------------------------------------------------------------
// Types / helpers
// ---------------------------------------------------------------------------
typedef __attribute__((ext_vector_type(16))) __bf16 bf16x16;
typedef __attribute__((ext_vector_type(8)))  float  f32x8;

union Frag { uint4 q[2]; bf16x16 v; };

__device__ __forceinline__ float bf2f(__bf16 b) {
    unsigned short u = __builtin_bit_cast(unsigned short, b);
    unsigned int x = ((unsigned int)u) << 16;
    return __builtin_bit_cast(float, x);
}
__device__ __forceinline__ __bf16 f2bf(float f) {
    unsigned int b = __builtin_bit_cast(unsigned int, f);
    unsigned int r = b + 0x7fffu + ((b >> 16) & 1u);   // round-to-nearest-even
    unsigned short h = (unsigned short)(r >> 16);
    return __builtin_bit_cast(__bf16, h);
}
__device__ __forceinline__ float sigmoidf_(float x) { return 1.f / (1.f + __expf(-x)); }
__device__ __forceinline__ float dswishf_(float x)  { return x * sigmoidf_(x - 1.f); }

// Problem constants
#define LSEQ 512
#define NB   16
#define DMODEL 512
#define HEADS 8
#define HD 64
#define PHD 32
#define FF 2048
#define TOK (LSEQ*NB)          // 8192
#define PPOS (2*LSEQ-1)        // 1023

// ---------------------------------------------------------------------------
// f32 -> bf16 convert
// ---------------------------------------------------------------------------
__global__ __launch_bounds__(256) void f32_to_bf16_kernel(const float* __restrict__ in,
                                                          __bf16* __restrict__ out, int n) {
    int i = blockIdx.x * 256 + threadIdx.x;
    if (i < n) out[i] = f2bf(in[i]);
}

// ---------------------------------------------------------------------------
// Generic tiled GEMM:  C[M,N] = A[M,K](bf16) @ W[N,K]^T(bf16) (+bias)(+res)(+act)
// Block = 256 threads (8 waves), tile 128x64, BK=32.
// Wave w: rows (w&3)*32, cols (w>>2)*32  => 2x2 WMMA 16x16 tiles.
// Tiles are staged global->LDS with GLOBAL_LOAD_ASYNC_TO_LDS_B128 (ASYNCcnt),
// double-buffered: issue tile t+1 while computing tile t.
// OOB rows are address-clamped: garbage affects only epilogue-masked C rows.
// ---------------------------------------------------------------------------
template <int ACT, bool HAS_BIAS, bool HAS_RES>
__global__ __launch_bounds__(256) void gemm_wmma_kernel(
    const __bf16* __restrict__ A, const __bf16* __restrict__ W,
    const float* __restrict__ bias, const float* __restrict__ res,
    float* __restrict__ outF, __bf16* __restrict__ outB,
    int M, int N, int K)
{
    __shared__ __align__(16) __bf16 As[2][128 * 32];
    __shared__ __align__(16) __bf16 Bs[2][64 * 32];

    const int tid  = threadIdx.x;
    const int lane = tid & 31;
    const int wave = tid >> 5;
    const int m0 = blockIdx.y * 128;
    const int n0 = blockIdx.x * 64;
    const int wr = wave & 3, wc = wave >> 2;
    const int l16 = lane & 15;
    const int lh  = (lane >= 16) ? 1 : 0;

    f32x8 acc[2][2] = {};

    // Issue one K-slab (3 async-to-LDS instructions per wave: 2 for A, 1 for W).
    auto issue_tile = [&](int k0, int buf) {
        #pragma unroll
        for (int i = 0; i < 2; ++i) {
            int idx = tid + i * 256;
            int r = idx >> 2, c = (idx & 3) * 8;
            int gr = m0 + r; if (gr >= M) gr = M - 1;   // clamp (epilogue masks)
            const __bf16* gp = A + (size_t)gr * K + k0 + c;
            unsigned lp = (unsigned)(uintptr_t)(&As[buf][r * 32 + c]);
            asm volatile("global_load_async_to_lds_b128 %0, %1, off"
                         :: "v"(lp), "v"(gp) : "memory");
        }
        {
            int r = tid >> 2, c = (tid & 3) * 8;
            int gn = n0 + r; if (gn >= N) gn = N - 1;
            const __bf16* gp = W + (size_t)gn * K + k0 + c;
            unsigned lp = (unsigned)(uintptr_t)(&Bs[buf][r * 32 + c]);
            asm volatile("global_load_async_to_lds_b128 %0, %1, off"
                         :: "v"(lp), "v"(gp) : "memory");
        }
    };

    const int T = K >> 5;   // K/32 slabs
    issue_tile(0, 0);

    for (int t = 0; t < T; ++t) {
        const int buf = t & 1;
        if (t + 1 < T) {
            issue_tile((t + 1) * 32, buf ^ 1);
            // wait for current tile only (3 newer async ops may stay in flight;
            // async loads complete in order)
            asm volatile("s_wait_asynccnt 0x3" ::: "memory");
        } else {
            asm volatile("s_wait_asynccnt 0x0" ::: "memory");
        }
        __syncthreads();

        Frag af[2], bf[2];
        #pragma unroll
        for (int mt = 0; mt < 2; ++mt) {
            int m = wr * 32 + mt * 16 + l16;
            int kb = lh ? 16 : 0;
            af[mt].q[0] = *(uint4*)(&As[buf][m * 32 + kb]);
            af[mt].q[1] = *(uint4*)(&As[buf][m * 32 + kb + 8]);
        }
        #pragma unroll
        for (int nt = 0; nt < 2; ++nt) {
            int n = wc * 32 + nt * 16 + l16;
            int kb = lh ? 16 : 0;
            bf[nt].q[0] = *(uint4*)(&Bs[buf][n * 32 + kb]);
            bf[nt].q[1] = *(uint4*)(&Bs[buf][n * 32 + kb + 8]);
        }
        #pragma unroll
        for (int mt = 0; mt < 2; ++mt)
            #pragma unroll
            for (int nt = 0; nt < 2; ++nt)
                acc[mt][nt] = __builtin_amdgcn_wmma_f32_16x16x32_bf16(
                    false, af[mt].v, false, bf[nt].v, (short)0, acc[mt][nt], false, false);
        __syncthreads();   // all waves done reading buf before it is re-filled
    }

    // Epilogue
    #pragma unroll
    for (int mt = 0; mt < 2; ++mt) {
        #pragma unroll
        for (int nt = 0; nt < 2; ++nt) {
            int n = n0 + wc * 32 + nt * 16 + l16;
            if (n >= N) continue;
            float bv = HAS_BIAS ? bias[n] : 0.f;
            #pragma unroll
            for (int v = 0; v < 8; ++v) {
                int m = m0 + wr * 32 + mt * 16 + v + (lh ? 8 : 0);
                if (m >= M) continue;
                float val = acc[mt][nt][v] + bv;
                if (HAS_RES) val += res[(size_t)m * N + n];
                if (ACT == 1) val = dswishf_(val);
                if (outF) outF[(size_t)m * N + n] = val;
                if (outB) outB[(size_t)m * N + n] = f2bf(val);
            }
        }
    }
}

// ---------------------------------------------------------------------------
// Repack kernels
// ---------------------------------------------------------------------------
__global__ __launch_bounds__(256) void repack_qkvp_kernel(
    const __bf16* __restrict__ proj, __bf16* __restrict__ qb, __bf16* __restrict__ kb,
    __bf16* __restrict__ vT, __bf16* __restrict__ pb)
{
    int idx = blockIdx.x * 256 + threadIdx.x;
    if (idx >= TOK * 1536) return;
    int row = idx / 1536, c = idx % 1536;
    int l = row >> 4, n = row & 15;
    __bf16 val = proj[idx];
    if (c < 512) {
        int h = c >> 6, d = c & 63;
        qb[(((size_t)(n * HEADS + h)) * LSEQ + l) * HD + d] = val;
    } else if (c < 1024) {
        int c2 = c - 512; int h = c2 >> 6, d = c2 & 63;
        kb[(((size_t)(n * HEADS + h)) * LSEQ + l) * HD + d] = val;
    } else if (c < 1280) {
        int c2 = c - 1024; int h = c2 >> 5, d = c2 & 31;
        vT[(((size_t)(n * HEADS + h)) * PHD + d) * LSEQ + l] = val;
    } else {
        int c2 = c - 1280; int h = c2 >> 5, d = c2 & 31;
        pb[(((size_t)(n * HEADS + h)) * LSEQ + l) * PHD + d] = val;
    }
}

__global__ __launch_bounds__(256) void repack_pos_kernel(
    const __bf16* __restrict__ posP, __bf16* __restrict__ posb)
{
    int idx = blockIdx.x * 256 + threadIdx.x;
    if (idx >= PPOS * 256) return;
    int j = idx >> 8, c = idx & 255;
    int h = c >> 5, d = c & 31;
    posb[((size_t)h * PPOS + j) * PHD + d] = posP[idx];
}

__global__ __launch_bounds__(256) void repack_v2_kernel(
    const __bf16* __restrict__ v2, __bf16* __restrict__ v2T)
{
    int idx = blockIdx.x * 256 + threadIdx.x;
    if (idx >= TOK * 256) return;
    int row = idx >> 8, c = idx & 255;
    int l = row >> 4, n = row & 15;
    int h = c >> 5, d = c & 31;
    v2T[(((size_t)(n * HEADS + h)) * PHD + d) * LSEQ + l] = v2[idx];
}

// ---------------------------------------------------------------------------
// Fused attention scores + rel-pos gather + softmax -> bf16 probs.
// One block per (n, h, 16-row l-block). 256 threads (8 waves).
// sc: 16x512 f32 scores; pos term computed as a 16x128 WMMA GEMM over the
// contiguous j-window [jbase, jbase+128) where j = L-1 + m - l.
// ---------------------------------------------------------------------------
__global__ __launch_bounds__(256) void attn_softmax_kernel(
    const __bf16* __restrict__ qb, const __bf16* __restrict__ kb,
    const __bf16* __restrict__ pb, const __bf16* __restrict__ posb,
    __bf16* __restrict__ attnB)
{
    __shared__ __align__(16) float  sc[16 * 512];
    __shared__ __align__(16) float  posf[16 * 128];
    __shared__ __align__(16) __bf16 qs[16 * 64];
    __shared__ __align__(16) __bf16 ps[16 * 32];
    __shared__ __align__(16) __bf16 ks[64 * 64];
    __shared__ __align__(16) __bf16 poss[128 * 32];

    const int tid = threadIdx.x, lane = tid & 31, wave = tid >> 5;
    const int l0 = blockIdx.x * 16, h = blockIdx.y, n = blockIdx.z;
    const int l16 = lane & 15;
    const int lh = (lane >= 16) ? 1 : 0;
    const size_t nh = (size_t)(n * HEADS + h);

    if (tid < 128) {  // q tile 16x64
        int r = tid >> 3, c = (tid & 7) * 8;
        *(uint4*)(qs + r * 64 + c) = *(const uint4*)(qb + (nh * LSEQ + l0 + r) * HD + c);
    }
    if (tid < 64) {   // p tile 16x32
        int r = tid >> 2, c = (tid & 3) * 8;
        *(uint4*)(ps + r * 32 + c) = *(const uint4*)(pb + (nh * LSEQ + l0 + r) * PHD + c);
    }
    __syncthreads();

    for (int mb = 0; mb < 8; ++mb) {
        const int jbase = (LSEQ - 1) + mb * 64 - l0 - 15;
        // k tile 64x64 (rows m, cols d) — 512 uint4
        #pragma unroll
        for (int i = 0; i < 2; ++i) {
            int idx = tid + i * 256;
            int r = idx >> 3, c = (idx & 7) * 8;
            *(uint4*)(ks + r * 64 + c) =
                *(const uint4*)(kb + (nh * LSEQ + mb * 64 + r) * HD + c);
        }
        // pos rows j in [jbase, jbase+128) zero-filled outside [0, PPOS) — 512 uint4
        #pragma unroll
        for (int i = 0; i < 2; ++i) {
            int idx = tid + i * 256;
            int r = idx >> 2, c = (idx & 3) * 8;
            int j = jbase + r;
            uint4 val = make_uint4(0u, 0u, 0u, 0u);
            if (j >= 0 && j < PPOS)
                val = *(const uint4*)(posb + ((size_t)h * PPOS + j) * PHD + c);
            *(uint4*)(poss + r * 32 + c) = val;
        }
        __syncthreads();

        // q @ k^T : 16x64 output, 4 col-tiles handled by waves 0..3, K=64
        if (wave < 4) {
            f32x8 s = {};
            #pragma unroll
            for (int ks0 = 0; ks0 < 64; ks0 += 32) {
                Frag a, b;
                int kba = lh ? 16 : 0;
                a.q[0] = *(uint4*)(qs + l16 * 64 + ks0 + kba);
                a.q[1] = *(uint4*)(qs + l16 * 64 + ks0 + kba + 8);
                int mm = wave * 16 + l16;
                int kbb = lh ? 16 : 0;
                b.q[0] = *(uint4*)(ks + mm * 64 + ks0 + kbb);
                b.q[1] = *(uint4*)(ks + mm * 64 + ks0 + kbb + 8);
                s = __builtin_amdgcn_wmma_f32_16x16x32_bf16(false, a.v, false, b.v,
                                                            (short)0, s, false, false);
            }
            #pragma unroll
            for (int v = 0; v < 8; ++v)
                sc[(v + (lh ? 8 : 0)) * 512 + mb * 64 + wave * 16 + l16] = s[v];
        }
        // p @ pos^T : 16x128 output, 8 col-tiles, one per wave, K=32
        {
            f32x8 s = {};
            Frag a, b;
            int kba = lh ? 16 : 0;
            a.q[0] = *(uint4*)(ps + l16 * 32 + kba);
            a.q[1] = *(uint4*)(ps + l16 * 32 + kba + 8);
            int jj = wave * 16 + l16;
            int kbb = lh ? 16 : 0;
            b.q[0] = *(uint4*)(poss + jj * 32 + kbb);
            b.q[1] = *(uint4*)(poss + jj * 32 + kbb + 8);
            s = __builtin_amdgcn_wmma_f32_16x16x32_bf16(false, a.v, false, b.v,
                                                        (short)0, s, false, false);
            #pragma unroll
            for (int v = 0; v < 8; ++v)
                posf[(v + (lh ? 8 : 0)) * 128 + jj] = s[v];
        }
        __syncthreads();
        // diagonal gather: scores[l', m'] += pos[l', m' - l' + 15]
        for (int t = tid; t < 16 * 64; t += 256) {
            int lp = t >> 6, mp = t & 63;
            sc[lp * 512 + mb * 64 + mp] += posf[lp * 128 + (mp - lp + 15)];
        }
        __syncthreads();
    }

    // softmax per row, write bf16 probs
    if (tid < 16) {
        float* row = sc + tid * 512;
        float mx = -3.0e38f;
        for (int m = 0; m < 512; ++m) mx = fmaxf(mx, row[m]);
        float sum = 0.f;
        for (int m = 0; m < 512; ++m) { float e = __expf(row[m] - mx); row[m] = e; sum += e; }
        float inv = 1.f / sum;
        __bf16* dst = attnB + (nh * LSEQ + l0 + tid) * LSEQ;
        for (int m = 0; m < 512; ++m) dst[m] = f2bf(row[m] * inv);
    }
}

// ---------------------------------------------------------------------------
// out[l, d] = attn[l, :] @ v[:, d]  per (n, h). Block: 64 rows x 32 cols.
// A = attn bf16 (row-major over m), B = vT (d-major rows, m contiguous).
// Output written straight into the (token-row, h*32+d) layout.
// ---------------------------------------------------------------------------
__global__ __launch_bounds__(256) void attn_av_kernel(
    const __bf16* __restrict__ attnB, const __bf16* __restrict__ vT,
    __bf16* __restrict__ outB)
{
    __shared__ __align__(16) __bf16 at[64 * 32];
    __shared__ __align__(16) __bf16 vb[32 * 32];

    const int tid = threadIdx.x, lane = tid & 31, wave = tid >> 5;
    const int l0 = blockIdx.x * 64, h = blockIdx.y, n = blockIdx.z;
    const int l16 = lane & 15;
    const int lh = (lane >= 16) ? 1 : 0;
    const int mt = wave & 3, nt = wave >> 2;
    const size_t nh = (size_t)(n * HEADS + h);

    f32x8 acc = {};
    for (int m0 = 0; m0 < LSEQ; m0 += 32) {
        {
            int r = tid >> 2, c = (tid & 3) * 8;
            *(uint4*)(at + r * 32 + c) =
                *(const uint4*)(attnB + (nh * LSEQ + l0 + r) * LSEQ + m0 + c);
        }
        if (tid < 128) {
            int r = tid >> 2, c = (tid & 3) * 8;
            *(uint4*)(vb + r * 32 + c) =
                *(const uint4*)(vT + (nh * PHD + r) * LSEQ + m0 + c);
        }
        __syncthreads();
        Frag a, b;
        int m = mt * 16 + l16;
        int kba = lh ? 16 : 0;
        a.q[0] = *(uint4*)(at + m * 32 + kba);
        a.q[1] = *(uint4*)(at + m * 32 + kba + 8);
        int d = nt * 16 + l16;
        int kbb = lh ? 16 : 0;
        b.q[0] = *(uint4*)(vb + d * 32 + kbb);
        b.q[1] = *(uint4*)(vb + d * 32 + kbb + 8);
        acc = __builtin_amdgcn_wmma_f32_16x16x32_bf16(false, a.v, false, b.v,
                                                      (short)0, acc, false, false);
        __syncthreads();
    }
    int d = nt * 16 + l16;
    #pragma unroll
    for (int v = 0; v < 8; ++v) {
        int l = l0 + mt * 16 + v + (lh ? 8 : 0);
        outB[((size_t)l * NB + n) * 256 + h * PHD + d] = f2bf(acc[v]);
    }
}

// ---------------------------------------------------------------------------
// GLU gate:  g = a * sigmoid(b)  with a,b = split(y1, 2, channel)
// ---------------------------------------------------------------------------
__global__ __launch_bounds__(256) void glu_kernel(const float* __restrict__ y1,
                                                  __bf16* __restrict__ gb)
{
    int idx = blockIdx.x * 256 + threadIdx.x;
    if (idx >= TOK * DMODEL) return;
    int row = idx >> 9, c = idx & 511;
    float a = y1[(size_t)row * 1024 + c];
    float b = y1[(size_t)row * 1024 + 512 + c];
    gb[idx] = f2bf(a * sigmoidf_(b));
}

// ---------------------------------------------------------------------------
// Depthwise conv K=31 along l (pad 15) + bias + double_swish
// ---------------------------------------------------------------------------
__global__ __launch_bounds__(256) void dwconv_kernel(
    const __bf16* __restrict__ gb, const float* __restrict__ w,
    const float* __restrict__ bias, __bf16* __restrict__ dwb)
{
    int idx = blockIdx.x * 256 + threadIdx.x;
    if (idx >= TOK * DMODEL) return;
    int c = idx & 511;
    int row = idx >> 9;
    int l = row >> 4, n = row & 15;
    float acc = bias[c];
    #pragma unroll
    for (int k = 0; k < 31; ++k) {
        int ll = l + k - 15;
        if (ll >= 0 && ll < LSEQ)
            acc += w[c * 31 + k] * bf2f(gb[(((size_t)ll * NB) + n) * DMODEL + c]);
    }
    dwb[idx] = f2bf(dswishf_(acc));
}

// ---------------------------------------------------------------------------
// Final balancer-style norm:  out = src * rsqrt(mean(src^2) + exp(eps))
// ---------------------------------------------------------------------------
__global__ __launch_bounds__(256) void rmsnorm_kernel(const float* __restrict__ src,
                                                      const float* __restrict__ epsp,
                                                      float* __restrict__ out)
{
    __shared__ float red[256];
    int row = blockIdx.x, tid = threadIdx.x;
    float s = 0.f;
    for (int c = tid; c < DMODEL; c += 256) {
        float v = src[(size_t)row * DMODEL + c];
        s += v * v;
    }
    red[tid] = s;
    __syncthreads();
    for (int st = 128; st > 0; st >>= 1) {
        if (tid < st) red[tid] += red[tid + st];
        __syncthreads();
    }
    float scale = rsqrtf(red[0] / (float)DMODEL + __expf(epsp[0]));
    for (int c = tid; c < DMODEL; c += 256)
        out[(size_t)row * DMODEL + c] = src[(size_t)row * DMODEL + c] * scale;
}

// ---------------------------------------------------------------------------
// Host-side orchestration
// ---------------------------------------------------------------------------
extern "C" void kernel_launch(void* const* d_in, const int* in_sizes, int n_in,
                              void* d_out, int out_size, void* d_ws, size_t ws_size,
                              hipStream_t stream)
{
    (void)in_sizes; (void)n_in; (void)out_size; (void)ws_size;
    const float* x          = (const float*)d_in[0];
    const float* pos_emb    = (const float*)d_in[1];
    const float* attn_in_w  = (const float*)d_in[2];
    const float* attn_in_b  = (const float*)d_in[3];
    const float* pos_w      = (const float*)d_in[4];
    const float* attn_out_w = (const float*)d_in[5];
    const float* attn_out_b = (const float*)d_in[6];
    const float* attn_in2_w = (const float*)d_in[7];
    const float* attn_out2_w= (const float*)d_in[8];
    const float* attn_out2_b= (const float*)d_in[9];
    const float* ff1_in_w   = (const float*)d_in[10];
    const float* ff1_in_b   = (const float*)d_in[11];
    const float* ff1_out_w  = (const float*)d_in[12];
    const float* ff1_out_b  = (const float*)d_in[13];
    const float* ff2_in_w   = (const float*)d_in[14];
    const float* ff2_in_b   = (const float*)d_in[15];
    const float* ff2_out_w  = (const float*)d_in[16];
    const float* ff2_out_b  = (const float*)d_in[17];
    const float* pw1_w      = (const float*)d_in[18];
    const float* pw1_b      = (const float*)d_in[19];
    const float* dw_w       = (const float*)d_in[20];
    const float* dw_b       = (const float*)d_in[21];
    const float* pw2_w      = (const float*)d_in[22];
    const float* pw2_b      = (const float*)d_in[23];
    const float* epsp       = (const float*)d_in[24];
    float* outp = (float*)d_out;

    // Arena allocator over d_ws
    char* base = (char*)d_ws;
    size_t cur = 0;
    auto alloc = [&](size_t bytes) -> char* {
        cur = (cur + 255) & ~(size_t)255;
        char* p = base + cur;
        cur += bytes;
        return p;
    };

    // bf16 weight copies
    __bf16* w_ff1_in   = (__bf16*)alloc((size_t)FF * DMODEL * 2);
    __bf16* w_ff1_out  = (__bf16*)alloc((size_t)DMODEL * FF * 2);
    __bf16* w_attn_in  = (__bf16*)alloc((size_t)1536 * DMODEL * 2);
    __bf16* w_pos      = (__bf16*)alloc((size_t)256 * DMODEL * 2);
    __bf16* w_attn_out = (__bf16*)alloc((size_t)DMODEL * 256 * 2);
    __bf16* w_attn_in2 = (__bf16*)alloc((size_t)256 * DMODEL * 2);
    __bf16* w_attn_out2= (__bf16*)alloc((size_t)DMODEL * 256 * 2);
    __bf16* w_ff2_in   = (__bf16*)alloc((size_t)FF * DMODEL * 2);
    __bf16* w_ff2_out  = (__bf16*)alloc((size_t)DMODEL * FF * 2);
    __bf16* w_pw1      = (__bf16*)alloc((size_t)1024 * DMODEL * 2);
    __bf16* w_pw2      = (__bf16*)alloc((size_t)DMODEL * DMODEL * 2);
    // activations
    __bf16* xb    = (__bf16*)alloc((size_t)TOK * DMODEL * 2);
    __bf16* peb   = (__bf16*)alloc((size_t)PPOS * DMODEL * 2);
    __bf16* hb    = (__bf16*)alloc((size_t)TOK * FF * 2);          // ff hidden (reused)
    float*  srcF  = (float*) alloc((size_t)TOK * DMODEL * 4);
    __bf16* srcb  = (__bf16*)alloc((size_t)TOK * DMODEL * 2);
    __bf16* projB = (__bf16*)alloc((size_t)TOK * 1536 * 2);
    __bf16* qb    = (__bf16*)alloc((size_t)NB * HEADS * LSEQ * HD * 2);
    __bf16* kb    = (__bf16*)alloc((size_t)NB * HEADS * LSEQ * HD * 2);
    __bf16* vT    = (__bf16*)alloc((size_t)NB * HEADS * PHD * LSEQ * 2);
    __bf16* pb    = (__bf16*)alloc((size_t)NB * HEADS * LSEQ * PHD * 2);
    __bf16* posPb = (__bf16*)alloc((size_t)PPOS * 256 * 2);
    __bf16* posb  = (__bf16*)alloc((size_t)HEADS * PPOS * PHD * 2);
    __bf16* attnB = (__bf16*)alloc((size_t)NB * HEADS * LSEQ * LSEQ * 2);
    __bf16* out1b = (__bf16*)alloc((size_t)TOK * 256 * 2);
    __bf16* v2b   = (__bf16*)alloc((size_t)TOK * 256 * 2);
    __bf16* v2T   = (__bf16*)alloc((size_t)NB * HEADS * PHD * LSEQ * 2);
    __bf16* out2b = (__bf16*)alloc((size_t)TOK * 256 * 2);
    float*  y1F   = (float*) alloc((size_t)TOK * 1024 * 4);
    __bf16* gb    = (__bf16*)alloc((size_t)TOK * DMODEL * 2);
    __bf16* dwb   = (__bf16*)alloc((size_t)TOK * DMODEL * 2);

    auto cvt = [&](const float* s, __bf16* d, int n) {
        f32_to_bf16_kernel<<<dim3((n + 255) / 256), dim3(256), 0, stream>>>(s, d, n);
    };
    auto gemm_grid = [](int M, int N) { return dim3((N + 63) / 64, (M + 127) / 128); };

    // --- weight / input conversions -------------------------------------
    cvt(ff1_in_w,   w_ff1_in,   FF * DMODEL);
    cvt(ff1_out_w,  w_ff1_out,  DMODEL * FF);
    cvt(attn_in_w,  w_attn_in,  1536 * DMODEL);
    cvt(pos_w,      w_pos,      256 * DMODEL);
    cvt(attn_out_w, w_attn_out, DMODEL * 256);
    cvt(attn_in2_w, w_attn_in2, 256 * DMODEL);
    cvt(attn_out2_w,w_attn_out2,DMODEL * 256);
    cvt(ff2_in_w,   w_ff2_in,   FF * DMODEL);
    cvt(ff2_out_w,  w_ff2_out,  DMODEL * FF);
    cvt(pw1_w,      w_pw1,      1024 * DMODEL);
    cvt(pw2_w,      w_pw2,      DMODEL * DMODEL);
    cvt(x,          xb,         TOK * DMODEL);
    cvt(pos_emb,    peb,        PPOS * DMODEL);

    // --- FF1: src = x + dswish(x@W1^T+b1)@W2^T+b2 ------------------------
    gemm_wmma_kernel<1, true, false><<<gemm_grid(TOK, FF), 256, 0, stream>>>(
        xb, w_ff1_in, ff1_in_b, nullptr, nullptr, hb, TOK, FF, DMODEL);
    gemm_wmma_kernel<0, true, true><<<gemm_grid(TOK, DMODEL), 256, 0, stream>>>(
        hb, w_ff1_out, ff1_out_b, x, srcF, srcb, TOK, DMODEL, FF);

    // --- attention projections ------------------------------------------
    gemm_wmma_kernel<0, true, false><<<gemm_grid(TOK, 1536), 256, 0, stream>>>(
        srcb, w_attn_in, attn_in_b, nullptr, nullptr, projB, TOK, 1536, DMODEL);
    repack_qkvp_kernel<<<dim3((TOK * 1536 + 255) / 256), dim3(256), 0, stream>>>(
        projB, qb, kb, vT, pb);
    gemm_wmma_kernel<0, false, false><<<gemm_grid(PPOS, 256), 256, 0, stream>>>(
        peb, w_pos, nullptr, nullptr, nullptr, posPb, PPOS, 256, DMODEL);
    repack_pos_kernel<<<dim3((PPOS * 256 + 255) / 256), dim3(256), 0, stream>>>(
        posPb, posb);

    // --- fused scores + rel-pos + softmax --------------------------------
    attn_softmax_kernel<<<dim3(LSEQ / 16, HEADS, NB), dim3(256), 0, stream>>>(
        qb, kb, pb, posb, attnB);

    // --- out = attn @ v ; src += out @ Wo^T + bo -------------------------
    attn_av_kernel<<<dim3(LSEQ / 64, HEADS, NB), dim3(256), 0, stream>>>(attnB, vT, out1b);
    gemm_wmma_kernel<0, true, true><<<gemm_grid(TOK, DMODEL), 256, 0, stream>>>(
        out1b, w_attn_out, attn_out_b, srcF, srcF, srcb, TOK, DMODEL, 256);

    // --- second value path -----------------------------------------------
    gemm_wmma_kernel<0, false, false><<<gemm_grid(TOK, 256), 256, 0, stream>>>(
        srcb, w_attn_in2, nullptr, nullptr, nullptr, v2b, TOK, 256, DMODEL);
    repack_v2_kernel<<<dim3((TOK * 256 + 255) / 256), dim3(256), 0, stream>>>(v2b, v2T);
    attn_av_kernel<<<dim3(LSEQ / 64, HEADS, NB), dim3(256), 0, stream>>>(attnB, v2T, out2b);
    gemm_wmma_kernel<0, true, true><<<gemm_grid(TOK, DMODEL), 256, 0, stream>>>(
        out2b, w_attn_out2, attn_out2_b, srcF, srcF, srcb, TOK, DMODEL, 256);

    // --- conv module -----------------------------------------------------
    gemm_wmma_kernel<0, true, false><<<gemm_grid(TOK, 1024), 256, 0, stream>>>(
        srcb, w_pw1, pw1_b, nullptr, y1F, nullptr, TOK, 1024, DMODEL);
    glu_kernel<<<dim3((TOK * DMODEL + 255) / 256), dim3(256), 0, stream>>>(y1F, gb);
    dwconv_kernel<<<dim3((TOK * DMODEL + 255) / 256), dim3(256), 0, stream>>>(
        gb, dw_w, dw_b, dwb);
    gemm_wmma_kernel<0, true, true><<<gemm_grid(TOK, DMODEL), 256, 0, stream>>>(
        dwb, w_pw2, pw2_b, srcF, srcF, srcb, TOK, DMODEL, DMODEL);

    // --- FF2 --------------------------------------------------------------
    gemm_wmma_kernel<1, true, false><<<gemm_grid(TOK, FF), 256, 0, stream>>>(
        srcb, w_ff2_in, ff2_in_b, nullptr, nullptr, hb, TOK, FF, DMODEL);
    gemm_wmma_kernel<0, true, true><<<gemm_grid(TOK, DMODEL), 256, 0, stream>>>(
        hb, w_ff2_out, ff2_out_b, srcF, srcF, nullptr, TOK, DMODEL, FF);

    // --- final norm -------------------------------------------------------
    rmsnorm_kernel<<<dim3(TOK), dim3(256), 0, stream>>>(srcF, epsp, outp);
}
